// PhysNetInteractionLayer_53223234732350
// MI455X (gfx1250) — matrix operand order repaired and checked by reading, not compiled, not run on hardware
//
#include <hip/hip_runtime.h>
#include <hip/hip_bf16.h>
#include <cstdint>

// ---------------------------------------------------------------------------
// PhysNet interaction layer for gfx1250 (MI455X), wave32 + WMMA bf16.
// N=50000 nodes, E=1.6M edges, K=64 rbf, F=128 features, R=3 residual blocks.
//
// Roofline: ~40 GFLOP total vs ~470 MB of HBM traffic (rbf stream dominates);
// gather/scatter targets (25.6 MB each) are L2-resident on the 192 MB L2.
// Bandwidth-bound -> all GEMMs ride v_wmma_f32_16x16x32_bf16 (f32 accum).
// ---------------------------------------------------------------------------

typedef __bf16 bf16_t;
typedef __attribute__((ext_vector_type(16))) __bf16 v16bf;
typedef __attribute__((ext_vector_type(8)))  __bf16 v8bf;
typedef __attribute__((ext_vector_type(8)))  float  v8f;

#define NNODES 50000
#define NEDGES 1600000
#define KRBF   64
#define FDIM   128

// Manual RNE float->bf16 — measured cheaper (static mix) than clang's
// software lowering of the __bf16 cast on this toolchain.
__device__ __forceinline__ bf16_t f2bf(float f) {
  union { float f; uint32_t u; } x; x.f = f;
  uint32_t r = (x.u + 0x7FFFu + ((x.u >> 16) & 1u)) >> 16;   // RNE
  uint16_t s = (uint16_t)r;
  bf16_t b;
  __builtin_memcpy(&b, &s, 2);
  return b;
}

__device__ __forceinline__ float softplus_f(float x) {
  return fmaxf(x, 0.0f) + log1pf(__expf(-fabsf(x)));
}

__device__ __forceinline__ v8f wmma_bf16(v16bf a, v16bf b, v8f c) {
  return __builtin_amdgcn_wmma_f32_16x16x32_bf16(false, a, false, b,
                                                 (short)0, c, false, false);
}

// A fragment (16x32 bf16) from a row-major LDS tile [16][128].
// ISA layout: lane<16 -> row=lane, K chunks {0..7, 16..23}; lane>=16 -> {8..15, 24..31}.
__device__ __forceinline__ v16bf load_A_lds(const bf16_t* tile, int kstep, int lane) {
  int row = lane & 15;
  int hi  = lane >> 4;
  int k0  = kstep * 32 + hi * 8;
  const bf16_t* p = tile + row * FDIM;
  v8bf a0 = *(const v8bf*)(p + k0);
  v8bf a1 = *(const v8bf*)(p + k0 + 16);
  v16bf r;
#pragma unroll
  for (int i = 0; i < 8; ++i) { r[i] = a0[i]; r[8 + i] = a1[i]; }
  return r;
}

// B fragment (32x16 bf16) from packed weights: for each (ktile, n) the 32
// K-values are contiguous; lane<16 takes K {0..7,16..23}, lane>=16 takes K {8..15,24..31}.
__device__ __forceinline__ v16bf load_B_packed(const bf16_t* wp, int kstep, int c, int lane) {
  int n    = c * 16 + (lane & 15);
  int half = lane >> 4;
  const bf16_t* p = wp + ((size_t)(kstep * FDIM + n) * 32) + half * 16;
  return *(const v16bf*)p;  // 32B contiguous
}

// ---------------------------------------------------------------------------
// Pack a KxN f32 weight matrix to the bf16 B-fragment layout.
// ---------------------------------------------------------------------------
__global__ void pack_w_kernel(const float* __restrict__ W, bf16_t* __restrict__ out,
                              int K, int N) {
  int i = blockIdx.x * blockDim.x + threadIdx.x;
  if (i >= K * N) return;
  int k  = i / N, n = i - k * N;
  int kt = k >> 5, kr = k & 31;
  out[((size_t)(kt * N + n)) * 32 + ((kr >> 3) & 1) * 16 + (kr & 7) + ((kr >> 4) << 3)] = f2bf(W[i]);
}

// ---------------------------------------------------------------------------
// Node pre: xa = softplus(x); m = xa@wi+bi; xjsrc = xa@wj+bj.
// One wave per 16-row stripe, 8 waves per block.
// ---------------------------------------------------------------------------
__global__ __launch_bounds__(256) void node_pre_kernel(
    const float* __restrict__ x,
    const bf16_t* __restrict__ wi_p, const float* __restrict__ bi,
    const bf16_t* __restrict__ wj_p, const float* __restrict__ bj,
    float* __restrict__ m_out, float* __restrict__ xjsrc) {
  __shared__ bf16_t smem[8][16 * FDIM];
  int lane = threadIdx.x & 31;
  int wv   = threadIdx.x >> 5;
  int tb   = (blockIdx.x * 8 + wv) * 16;
  if (tb >= NNODES) return;
  bf16_t* tile = smem[wv];

  const float4* xp = (const float4*)(x + (size_t)tb * FDIM);
#pragma unroll
  for (int t = 0; t < 16; ++t) {
    int i = t * 32 + lane;
    float4 v = xp[i];
    bf16_t* d = tile + i * 4;
    d[0] = f2bf(softplus_f(v.x));
    d[1] = f2bf(softplus_f(v.y));
    d[2] = f2bf(softplus_f(v.z));
    d[3] = f2bf(softplus_f(v.w));
  }
  asm volatile("s_wait_dscnt 0" ::: "memory");

  v16bf A[4];
#pragma unroll
  for (int ks = 0; ks < 4; ++ks) A[ks] = load_A_lds(tile, ks, lane);

  int row = lane & 15, hi = lane >> 4;
#pragma unroll 1
  for (int w = 0; w < 2; ++w) {
    const bf16_t* wp  = w ? wj_p : wi_p;
    const float* bias = w ? bj : bi;
    float* out        = w ? xjsrc : m_out;
#pragma unroll 1
    for (int c = 0; c < 8; ++c) {
      v8f acc = {0.f, 0.f, 0.f, 0.f, 0.f, 0.f, 0.f, 0.f};
#pragma unroll
      for (int ks = 0; ks < 4; ++ks)
        acc = wmma_bf16(A[ks], load_B_packed(wp, ks, c, lane), acc);
      int col = c * 16 + row;
      float b = bias[col];
#pragma unroll
      for (int v = 0; v < 8; ++v)
        out[(size_t)(tb + v + hi * 8) * FDIM + col] = acc[v] + b;
    }
  }
}

// ---------------------------------------------------------------------------
// Edge: g = rbf@w_k2f; msgs = g * xjsrc[idx_j]; m[idx_i] += msgs (atomics).
// One wave per 16-edge tile. xjsrc & m are L2-resident (25.6MB each).
// ---------------------------------------------------------------------------
__global__ __launch_bounds__(256) void edge_kernel(
    const float* __restrict__ rbf,
    const int* __restrict__ idx_i, const int* __restrict__ idx_j,
    const bf16_t* __restrict__ wk_p,
    const float* __restrict__ xjsrc,
    float* __restrict__ m) {
  int lane = threadIdx.x & 31;
  int wave = (blockIdx.x * blockDim.x + threadIdx.x) >> 5;
  int tb   = wave * 16;
  if (tb >= NEDGES) return;
  int row = lane & 15, hi = lane >> 4;

  // A fragments straight from the f32 rbf stream (row = one edge, K=64).
  const float* rp = rbf + (size_t)(tb + row) * KRBF;
  v16bf A[2];
#pragma unroll
  for (int ks = 0; ks < 2; ++ks) {
    int k0 = ks * 32 + hi * 8;
    const float4* p0 = (const float4*)(rp + k0);
    const float4* p1 = (const float4*)(rp + k0 + 16);
    float4 a = p0[0], b = p0[1], c0 = p1[0], d = p1[1];
    float f[16] = {a.x, a.y, a.z, a.w, b.x, b.y, b.z, b.w,
                   c0.x, c0.y, c0.z, c0.w, d.x, d.y, d.z, d.w};
    v16bf r;
#pragma unroll
    for (int i = 0; i < 16; ++i) r[i] = f2bf(f[i]);
    A[ks] = r;
  }

  // Per-lane edge rows this lane's C elements map to: M = v + hi*8.
  int ir[8], jr[8];
#pragma unroll
  for (int v = 0; v < 8; ++v) {
    int e = tb + v + hi * 8;
    ir[v] = idx_i[e];
    jr[v] = idx_j[e];
  }

#pragma unroll 1
  for (int c = 0; c < 8; ++c) {
    v8f acc = {0.f, 0.f, 0.f, 0.f, 0.f, 0.f, 0.f, 0.f};
    acc = wmma_bf16(A[0], load_B_packed(wk_p, 0, c, lane), acc);
    acc = wmma_bf16(A[1], load_B_packed(wk_p, 1, c, lane), acc);
    int col = c * 16 + row;
#pragma unroll
    for (int v = 0; v < 8; ++v) {
      float val = acc[v] * xjsrc[(size_t)jr[v] * FDIM + col];
      unsafeAtomicAdd(&m[(size_t)ir[v] * FDIM + col], val);
    }
  }
}

// ---------------------------------------------------------------------------
// One residual block, fully fused per 16-row stripe:
//   y = softplus(m); m += (y@w1+b1)@w2 + b2
// ---------------------------------------------------------------------------
__global__ __launch_bounds__(256) void residual_kernel(
    float* __restrict__ m,
    const bf16_t* __restrict__ w1_p, const float* __restrict__ b1,
    const bf16_t* __restrict__ w2_p, const float* __restrict__ b2) {
  __shared__ bf16_t smem[8][16 * FDIM];
  int lane = threadIdx.x & 31;
  int wv   = threadIdx.x >> 5;
  int tb   = (blockIdx.x * 8 + wv) * 16;
  if (tb >= NNODES) return;
  bf16_t* tile = smem[wv];

  const float4* mp = (const float4*)(m + (size_t)tb * FDIM);
#pragma unroll
  for (int t = 0; t < 16; ++t) {
    int i = t * 32 + lane;
    float4 v = mp[i];
    bf16_t* d = tile + i * 4;
    d[0] = f2bf(softplus_f(v.x));
    d[1] = f2bf(softplus_f(v.y));
    d[2] = f2bf(softplus_f(v.z));
    d[3] = f2bf(softplus_f(v.w));
  }
  asm volatile("s_wait_dscnt 0" ::: "memory");

  v16bf A[4];
#pragma unroll
  for (int ks = 0; ks < 4; ++ks) A[ks] = load_A_lds(tile, ks, lane);
  int row = lane & 15, hi = lane >> 4;

  // GEMM1: t = y@w1 + b1  (keep whole stripe of t in registers)
  v8f accs[8];
#pragma unroll 1
  for (int c = 0; c < 8; ++c) {
    v8f acc = {0.f, 0.f, 0.f, 0.f, 0.f, 0.f, 0.f, 0.f};
#pragma unroll
    for (int ks = 0; ks < 4; ++ks)
      acc = wmma_bf16(A[ks], load_B_packed(w1_p, ks, c, lane), acc);
    float b = b1[c * 16 + row];
#pragma unroll
    for (int v = 0; v < 8; ++v) acc[v] += b;
    accs[c] = acc;
  }

  // Transpose hop through LDS: C-layout -> row-major bf16 tile for A of GEMM2.
#pragma unroll
  for (int c = 0; c < 8; ++c) {
    int col = c * 16 + row;
#pragma unroll
    for (int v = 0; v < 8; ++v)
      tile[(v + hi * 8) * FDIM + col] = f2bf(accs[c][v]);
  }
  asm volatile("s_wait_dscnt 0" ::: "memory");

#pragma unroll
  for (int ks = 0; ks < 4; ++ks) A[ks] = load_A_lds(tile, ks, lane);

  // GEMM2 + residual add
#pragma unroll 1
  for (int c = 0; c < 8; ++c) {
    v8f acc = {0.f, 0.f, 0.f, 0.f, 0.f, 0.f, 0.f, 0.f};
#pragma unroll
    for (int ks = 0; ks < 4; ++ks)
      acc = wmma_bf16(A[ks], load_B_packed(w2_p, ks, c, lane), acc);
    int col = c * 16 + row;
    float b = b2[col];
#pragma unroll
    for (int v = 0; v < 8; ++v) {
      size_t o = (size_t)(tb + v + hi * 8) * FDIM + col;
      m[o] = m[o] + acc[v] + b;
    }
  }
}

// ---------------------------------------------------------------------------
// Final: out = u*x + softplus(m)@w_out + b_out
// ---------------------------------------------------------------------------
__global__ __launch_bounds__(256) void final_kernel(
    const float* __restrict__ m, const float* __restrict__ x,
    const bf16_t* __restrict__ wo_p, const float* __restrict__ b_out,
    const float* __restrict__ u, float* __restrict__ out) {
  __shared__ bf16_t smem[8][16 * FDIM];
  int lane = threadIdx.x & 31;
  int wv   = threadIdx.x >> 5;
  int tb   = (blockIdx.x * 8 + wv) * 16;
  if (tb >= NNODES) return;
  bf16_t* tile = smem[wv];

  const float4* mp = (const float4*)(m + (size_t)tb * FDIM);
#pragma unroll
  for (int t = 0; t < 16; ++t) {
    int i = t * 32 + lane;
    float4 v = mp[i];
    bf16_t* d = tile + i * 4;
    d[0] = f2bf(softplus_f(v.x));
    d[1] = f2bf(softplus_f(v.y));
    d[2] = f2bf(softplus_f(v.z));
    d[3] = f2bf(softplus_f(v.w));
  }
  asm volatile("s_wait_dscnt 0" ::: "memory");

  v16bf A[4];
#pragma unroll
  for (int ks = 0; ks < 4; ++ks) A[ks] = load_A_lds(tile, ks, lane);
  int row = lane & 15, hi = lane >> 4;

#pragma unroll 1
  for (int c = 0; c < 8; ++c) {
    v8f acc = {0.f, 0.f, 0.f, 0.f, 0.f, 0.f, 0.f, 0.f};
#pragma unroll
    for (int ks = 0; ks < 4; ++ks)
      acc = wmma_bf16(A[ks], load_B_packed(wo_p, ks, c, lane), acc);
    int col = c * 16 + row;
    float bo = b_out[col];
    float uc = u[col];
#pragma unroll
    for (int v = 0; v < 8; ++v) {
      size_t o = (size_t)(tb + v + hi * 8) * FDIM + col;
      out[o] = uc * x[o] + acc[v] + bo;
    }
  }
}

// ---------------------------------------------------------------------------
extern "C" void kernel_launch(void* const* d_in, const int* in_sizes, int n_in,
                              void* d_out, int out_size, void* d_ws, size_t ws_size,
                              hipStream_t stream) {
  const float* x     = (const float*)d_in[0];
  const float* rbf   = (const float*)d_in[1];
  const int*   idx_i = (const int*)d_in[2];
  const int*   idx_j = (const int*)d_in[3];
  const float* w_k2f = (const float*)d_in[4];
  const float* wi    = (const float*)d_in[5];
  const float* bi    = (const float*)d_in[6];
  const float* wj    = (const float*)d_in[7];
  const float* bj    = (const float*)d_in[8];
  const float* rw1   = (const float*)d_in[9];
  const float* rb1   = (const float*)d_in[10];
  const float* rw2   = (const float*)d_in[11];
  const float* rb2   = (const float*)d_in[12];
  const float* w_out = (const float*)d_in[13];
  const float* b_out = (const float*)d_in[14];
  const float* u     = (const float*)d_in[15];
  float* out = (float*)d_out;
  (void)in_sizes; (void)n_in; (void)out_size; (void)ws_size;

  char* ws   = (char*)d_ws;
  size_t off = 0;
  auto alloc = [&](size_t bytes) -> void* {
    void* p = ws + off;
    off = (off + bytes + 255) & ~(size_t)255;
    return p;
  };

  float*  m_buf = (float*)alloc((size_t)NNODES * FDIM * sizeof(float));
  float*  xjsrc = (float*)alloc((size_t)NNODES * FDIM * sizeof(float));
  bf16_t* wk_p  = (bf16_t*)alloc((size_t)KRBF * FDIM * sizeof(bf16_t));
  bf16_t* wi_p  = (bf16_t*)alloc((size_t)FDIM * FDIM * sizeof(bf16_t));
  bf16_t* wj_p  = (bf16_t*)alloc((size_t)FDIM * FDIM * sizeof(bf16_t));
  bf16_t* w1_p[3], *w2_p[3];
  for (int r = 0; r < 3; ++r) {
    w1_p[r] = (bf16_t*)alloc((size_t)FDIM * FDIM * sizeof(bf16_t));
    w2_p[r] = (bf16_t*)alloc((size_t)FDIM * FDIM * sizeof(bf16_t));
  }
  bf16_t* wo_p = (bf16_t*)alloc((size_t)FDIM * FDIM * sizeof(bf16_t));

  const int TP  = 256;
  const int sqN = FDIM * FDIM;
  pack_w_kernel<<<(KRBF * FDIM + TP - 1) / TP, TP, 0, stream>>>(w_k2f, wk_p, KRBF, FDIM);
  pack_w_kernel<<<(sqN + TP - 1) / TP, TP, 0, stream>>>(wi, wi_p, FDIM, FDIM);
  pack_w_kernel<<<(sqN + TP - 1) / TP, TP, 0, stream>>>(wj, wj_p, FDIM, FDIM);
  for (int r = 0; r < 3; ++r) {
    pack_w_kernel<<<(sqN + TP - 1) / TP, TP, 0, stream>>>(rw1 + (size_t)r * sqN, w1_p[r], FDIM, FDIM);
    pack_w_kernel<<<(sqN + TP - 1) / TP, TP, 0, stream>>>(rw2 + (size_t)r * sqN, w2_p[r], FDIM, FDIM);
  }
  pack_w_kernel<<<(sqN + TP - 1) / TP, TP, 0, stream>>>(w_out, wo_p, FDIM, FDIM);

  const int nodeStripes = NNODES / 16;                 // 3125
  const int nodeBlocks  = (nodeStripes + 7) / 8;       // 391
  const int edgeTiles   = NEDGES / 16;                 // 100000
  const int edgeBlocks  = (edgeTiles + 7) / 8;         // 12500

  node_pre_kernel<<<nodeBlocks, 256, 0, stream>>>(x, wi_p, bi, wj_p, bj, m_buf, xjsrc);
  edge_kernel<<<edgeBlocks, 256, 0, stream>>>(rbf, idx_i, idx_j, wk_p, xjsrc, m_buf);
  for (int r = 0; r < 3; ++r)
    residual_kernel<<<nodeBlocks, 256, 0, stream>>>(m_buf, w1_p[r], rb1 + r * FDIM,
                                                    w2_p[r], rb2 + r * FDIM);
  final_kernel<<<nodeBlocks, 256, 0, stream>>>(m_buf, x, wo_p, b_out, u, out);
}
